// MultiHeadAttention_53566832115914
// MI455X (gfx1250) — matrix operand order repaired and checked by reading, not compiled, not run on hardware
//
#include <hip/hip_runtime.h>
#include <hip/hip_fp16.h>

typedef __attribute__((ext_vector_type(16))) _Float16 v16h;
typedef __attribute__((ext_vector_type(8)))  _Float16 v8h;
typedef __attribute__((ext_vector_type(8)))  float    v8f;
typedef __attribute__((ext_vector_type(4)))  float    v4f;
typedef __attribute__((ext_vector_type(4)))  int      v4i;

#define WMMA_F16(a, b, c) \
  __builtin_amdgcn_wmma_f32_16x16x32_f16(false, (a), false, (b), (short)0, (c), false, false)

constexpr int Bd  = 2;
constexpr int Sd  = 2048;
constexpr int Dd  = 1024;
constexpr int Hd  = 16;
constexpr int DKd = 64;
constexpr int Rr  = Bd * Sd;          // 4096 flattened rows

// ---- LDS layout for the attention kernel ---------------------------------
constexpr int SP          = 2064;                   // score row pitch (dwords), pads banks
constexpr int SCORES_OFF  = 0;
constexpr int SCORES_B    = 16 * SP * 4;            // 132096
constexpr int RED_OFF     = SCORES_OFF + SCORES_B;  // 16x8 f32
constexpr int ROWMAX_OFF  = RED_OFF + 512;
constexpr int RCP_OFF     = ROWMAX_OFF + 64;
constexpr int OBUF_OFF    = RCP_OFF + 64;           // 4x16x64 f32
constexpr int MBITS_OFF   = OBUF_OFF + 4 * 16 * 64 * 4;  // 16 rows x 64 words
constexpr int SMEM_TOTAL  = MBITS_OFF + 16 * 64 * 4;     // 153216 bytes

// ---- fragment loaders (all-f16 sources) -----------------------------------
// A-matrix (16xK rows): lane holds row (lane&15);
// element j -> K = kb + j (j<8) or kb + j + 8 (j>=8), kb = k0 + (lane>=16)*8.
__device__ inline v16h a_frag_f16(const _Float16* base, int stride, int m0, int k0, int lane) {
  int row = m0 + (lane & 15);
  int kb  = k0 + ((lane >> 4) << 3);
  const _Float16* p = base + (size_t)row * stride + kb;
  v8h x0 = *(const v8h*)(p);
  v8h x1 = *(const v8h*)(p + 16);
  v16h r;
#pragma unroll
  for (int i = 0; i < 8; ++i) { r[i] = x0[i]; r[8 + i] = x1[i]; }
  return r;
}

// B-matrix (Kx16 cols): lane holds column (lane&15);
// element j -> K = kb + j (j<8) or kb + j + 8 (j>=8), kb = k0 + (lane>=16)*8.
__device__ inline v16h b_frag_f16(const _Float16* base, int stride, int n0, int k0, int lane) {
  int col = n0 + (lane & 15);
  int kb  = k0 + ((lane >> 4) << 3);
  const _Float16* p = base + (size_t)col * stride + kb;
  v8h x0 = *(const v8h*)(p);
  v8h x1 = *(const v8h*)(p + 16);
  v16h r;
#pragma unroll
  for (int i = 0; i < 8; ++i) { r[i] = x0[i]; r[8 + i] = x1[i]; }
  return r;
}

// ---- Kernel 0: one-shot fp32 -> f16 conversion (8 elems/thread) ----------
__global__ void cvt_f32_to_f16_kernel(const float* __restrict__ src,
                                      _Float16* __restrict__ dst, int n) {
  int i = (blockIdx.x * blockDim.x + threadIdx.x) * 8;
  if (i >= n) return;
  v4f x0 = *(const v4f*)(src + i);
  v4f x1 = *(const v4f*)(src + i + 4);
  v8h r;
#pragma unroll
  for (int j = 0; j < 4; ++j) { r[j] = (_Float16)x0[j]; r[4 + j] = (_Float16)x1[j]; }
  *(v8h*)(dst + i) = r;
}

// ---- Kernel 1: fused QKV projection  (out = X @ W.T), 32x64 wave tiles ----
// grid.y: 0=Q, 1=K, 2=V.  Q,K stored [B,H,S,DK]; V stored transposed [B,H,DK,S].
__global__ void qkv_proj_kernel(const _Float16* __restrict__ Xq, const _Float16* __restrict__ Xk,
                                const _Float16* __restrict__ Xv,
                                const _Float16* __restrict__ Wqh, const _Float16* __restrict__ Wkh,
                                const _Float16* __restrict__ Wvh,
                                _Float16* __restrict__ Qh, _Float16* __restrict__ Kh,
                                _Float16* __restrict__ Vt) {
  int mode = blockIdx.y;
  const _Float16* X = (mode == 0) ? Xq  : (mode == 1) ? Xk  : Xv;
  const _Float16* W = (mode == 0) ? Wqh : (mode == 1) ? Wkh : Wvh;

  int lane = threadIdx.x & 31;
  int wave = threadIdx.x >> 5;
  int tile = blockIdx.x * 4 + wave;      // 0..2047 : 128 m-tiles(32) x 16 n-tiles(64)
  int m0 = (tile >> 4) * 32;
  int n0 = (tile & 15) * 64;

  v8f acc[8] = {};
  for (int k0 = 0; k0 < Dd; k0 += 32) {
    v16h a0 = a_frag_f16(X, Dd, m0,      k0, lane);
    v16h a1 = a_frag_f16(X, Dd, m0 + 16, k0, lane);
#pragma unroll
    for (int nt = 0; nt < 4; ++nt) {
      v16h bf = b_frag_f16(W, Dd, n0 + nt * 16, k0, lane);
      acc[nt]     = WMMA_F16(a0, bf, acc[nt]);
      acc[4 + nt] = WMMA_F16(a1, bf, acc[4 + nt]);
    }
  }

  int mrow = (lane >> 4) << 3;
  int ncol = lane & 15;
#pragma unroll
  for (int half = 0; half < 2; ++half) {
#pragma unroll
    for (int nt = 0; nt < 4; ++nt) {
#pragma unroll
      for (int r = 0; r < 8; ++r) {
        int mg = m0 + half * 16 + mrow + r;  // flattened b*S + s (tiles never cross batch)
        int e  = n0 + nt * 16 + ncol;        // feature index
        int b  = mg / Sd, s = mg % Sd;
        int h  = e / DKd, dk = e % DKd;
        _Float16 val = (_Float16)acc[half * 4 + nt][r];
        if (mode == 0)      Qh[(((size_t)(b * Hd + h)) * Sd + s) * DKd + dk] = val;
        else if (mode == 1) Kh[(((size_t)(b * Hd + h)) * Sd + s) * DKd + dk] = val;
        else                Vt[(((size_t)(b * Hd + h)) * DKd + dk) * Sd + s] = val;
      }
    }
  }
}

// ---- Kernel 2: attention (scores -> softmax -> weights out -> P@V) -------
// One WG (4 waves) per (b,h, 16-query tile). Score strip lives in LDS.
__global__ void attention_kernel(const _Float16* __restrict__ Qh, const _Float16* __restrict__ Kh,
                                 const _Float16* __restrict__ Vt, const int* __restrict__ mask,
                                 float* __restrict__ attnw, _Float16* __restrict__ AO) {
  extern __shared__ char smem[];
  float*    scores = (float*)(smem + SCORES_OFF);   // [16][SP]
  float*    red    = (float*)(smem + RED_OFF);      // [16][8]
  float*    rowmax = (float*)(smem + ROWMAX_OFF);   // [16]
  float*    rowrcp = (float*)(smem + RCP_OFF);      // [16]
  float*    obuf   = (float*)(smem + OBUF_OFF);     // [4][16][64]
  unsigned* mbits  = (unsigned*)(smem + MBITS_OFF); // [16][64] bit-packed mask

  int t    = threadIdx.x;
  int lane = t & 31;
  int wave = t >> 5;
  int bh = blockIdx.x >> 7;                       // 128 query tiles per head
  int qt = blockIdx.x & 127;
  int q0 = qt * 16;
  int b  = bh >> 4, h = bh & 15;

  const _Float16* Qbase = Qh + (size_t)bh * Sd * DKd;
  const _Float16* Kbase = Kh + (size_t)bh * Sd * DKd;
  const _Float16* Vbase = Vt + (size_t)bh * DKd * Sd;

  // Q A-fragments (held for the whole kernel): k-steps over DK=64
  v16h qf0 = a_frag_f16(Qbase, DKd, q0, 0,  lane);
  v16h qf1 = a_frag_f16(Qbase, DKd, q0, 32, lane);

  // Phase 0: bit-pack the mask strip into LDS (coalesced int4 reads)
  for (int w = t; w < 16 * 64; w += 128) {
    int row = w >> 6;
    int kw  = (w & 63) << 5;
    const int* mp = mask + ((size_t)(b * Sd + q0 + row)) * Sd + kw;
    unsigned bits = 0;
#pragma unroll
    for (int i = 0; i < 8; ++i) {
      v4i m4 = *(const v4i*)(mp + i * 4);
#pragma unroll
      for (int j = 0; j < 4; ++j) bits |= (m4[j] ? 1u : 0u) << (i * 4 + j);
    }
    mbits[w] = bits;
  }
  __syncthreads();

  int wkey0 = wave * 512;                         // this wave's key strip
  int mrow  = (lane >> 4) << 3;
  int ncol  = lane & 15;

  // Phase 1: scores = (Q K^T) * 1/sqrt(DK), masked -> LDS
  for (int kt = 0; kt < 32; ++kt) {
    int key0 = wkey0 + kt * 16;
    v8f acc = {};
    v16h b0 = b_frag_f16(Kbase, DKd, key0, 0,  lane);   // B = K^T (col = key row)
    acc = WMMA_F16(qf0, b0, acc);
    v16h b1 = b_frag_f16(Kbase, DKd, key0, 32, lane);
    acc = WMMA_F16(qf1, b1, acc);
    int key = key0 + ncol;
#pragma unroll
    for (int r = 0; r < 8; ++r) {
      int m  = mrow + r;
      int mk = (mbits[(m << 6) + (key >> 5)] >> (key & 31)) & 1;
      scores[m * SP + key] = mk ? acc[r] * 0.125f : -1e9f;
    }
  }
  __syncthreads();

  // Phase 2a: row max (8 threads per row, 256 elems each)
  {
    int row = t >> 3, seg = t & 7;
    float mx = -3.0e38f;
    const float* sr = scores + row * SP + seg * 256;
    for (int i = 0; i < 256; ++i) mx = fmaxf(mx, sr[i]);
    red[row * 8 + seg] = mx;
  }
  __syncthreads();
  if (t < 16) {
    float mx = red[t * 8];
    for (int i = 1; i < 8; ++i) mx = fmaxf(mx, red[t * 8 + i]);
    rowmax[t] = mx;
  }
  __syncthreads();

  // Phase 2b: row sum of exp
  {
    int row = t >> 3, seg = t & 7;
    float mx = rowmax[row], sm = 0.f;
    const float* sr = scores + row * SP + seg * 256;
    for (int i = 0; i < 256; ++i) sm += __expf(sr[i] - mx);
    red[row * 8 + seg] = sm;
  }
  __syncthreads();
  if (t < 16) {
    float sm = 0.f;
    for (int i = 0; i < 8; ++i) sm += red[t * 8 + i];
    rowrcp[t] = 1.0f / sm;
  }
  __syncthreads();

  // Phase 3: stream normalized weights to HBM AND write p back into LDS
  float* awbase = attnw + ((size_t)bh * Sd + q0) * Sd;
  for (int row = 0; row < 16; ++row) {
    float mx = rowmax[row], rc = rowrcp[row];
    float* sr  = scores + row * SP;
    float* dst = awbase + (size_t)row * Sd;
    for (int i = 0; i < 16; ++i) {
      int k = i * 128 + t;
      float p = __expf(sr[k] - mx) * rc;
      dst[k] = p;
      sr[k]  = p;
    }
  }
  __syncthreads();

  // Phase 4: O_partial = P @ V over this wave's 512 keys (16 k-steps of 32)
  v8f oacc[4] = {};
  int pm = lane & 15;
  for (int kt = 0; kt < 16; ++kt) {
    int kb    = wkey0 + kt * 32;
    int kbase = kb + ((lane >> 4) << 3);
    v16h pf;
#pragma unroll
    for (int j = 0; j < 16; ++j) {
      int k = kbase + j + ((j >> 3) << 3);          // j<8: +j ; j>=8: +j+8
      pf[j] = (_Float16)scores[pm * SP + k];
    }
#pragma unroll
    for (int nt = 0; nt < 4; ++nt) {
      v16h vf = b_frag_f16(Vbase, Sd, nt * 16, kb, lane);  // col = dk row of V^T
      oacc[nt] = WMMA_F16(pf, vf, oacc[nt]);
    }
  }

  // Phase 5: reduce partials across the 4 waves via LDS, emit f16 context
#pragma unroll
  for (int nt = 0; nt < 4; ++nt)
#pragma unroll
    for (int r = 0; r < 8; ++r) {
      int mm = mrow + r;
      int dk = nt * 16 + ncol;
      obuf[(wave * 16 + mm) * 64 + dk] = oacc[nt][r];
    }
  __syncthreads();
  {
    int row = t >> 3;
    int dkb = (t & 7) * 8;
    _Float16* dst = AO + ((size_t)(b * Sd + q0 + row)) * Dd + h * DKd + dkb;
    for (int i = 0; i < 8; ++i) {
      int dk  = dkb + i;
      float s = obuf[(0 * 16 + row) * 64 + dk] + obuf[(1 * 16 + row) * 64 + dk] +
                obuf[(2 * 16 + row) * 64 + dk] + obuf[(3 * 16 + row) * 64 + dk];
      dst[i] = (_Float16)s;
    }
  }
}

// ---- Kernel 3: output projection  out = AO @ Wo.T + bo  (32x64 tiles) ----
__global__ void out_proj_kernel(const _Float16* __restrict__ AO, const _Float16* __restrict__ Woh,
                                const float* __restrict__ bo, float* __restrict__ out) {
  int lane = threadIdx.x & 31;
  int wave = threadIdx.x >> 5;
  int tile = blockIdx.x * 4 + wave;      // 0..2047
  int m0 = (tile >> 4) * 32;
  int n0 = (tile & 15) * 64;

  v8f acc[8] = {};
  for (int k0 = 0; k0 < Dd; k0 += 32) {
    v16h a0 = a_frag_f16(AO, Dd, m0,      k0, lane);
    v16h a1 = a_frag_f16(AO, Dd, m0 + 16, k0, lane);
#pragma unroll
    for (int nt = 0; nt < 4; ++nt) {
      v16h bf = b_frag_f16(Woh, Dd, n0 + nt * 16, k0, lane);
      acc[nt]     = WMMA_F16(a0, bf, acc[nt]);
      acc[4 + nt] = WMMA_F16(a1, bf, acc[4 + nt]);
    }
  }

  int mrow = (lane >> 4) << 3;
  int ncol = lane & 15;
#pragma unroll
  for (int half = 0; half < 2; ++half) {
#pragma unroll
    for (int nt = 0; nt < 4; ++nt) {
#pragma unroll
      for (int r = 0; r < 8; ++r) {
        int mg = m0 + half * 16 + mrow + r;
        int e  = n0 + nt * 16 + ncol;
        out[(size_t)mg * Dd + e] = acc[half * 4 + nt][r] + bo[e];
      }
    }
  }
}

// ---- host-side launcher ---------------------------------------------------
extern "C" void kernel_launch(void* const* d_in, const int* in_sizes, int n_in,
                              void* d_out, int out_size, void* d_ws, size_t ws_size,
                              hipStream_t stream) {
  (void)in_sizes; (void)n_in; (void)out_size; (void)ws_size;
  const float* query = (const float*)d_in[0];
  const float* key   = (const float*)d_in[1];
  const float* value = (const float*)d_in[2];
  const int*   mask  = (const int*)d_in[3];
  const float* Wq    = (const float*)d_in[4];
  const float* Wk    = (const float*)d_in[5];
  const float* Wv    = (const float*)d_in[6];
  const float* Wo    = (const float*)d_in[7];
  const float* bo    = (const float*)d_in[8];

  float* out   = (float*)d_out;                       // [B,S,D]
  float* attnw = out + (size_t)Bd * Sd * Dd;          // [B,H,S,S]

  const size_t xElems = (size_t)Rr * Dd;              // 4,194,304
  const size_t wElems = (size_t)Dd * Dd;              // 1,048,576
  _Float16* Xqh = (_Float16*)d_ws;
  _Float16* Xkh = Xqh + xElems;
  _Float16* Xvh = Xkh + xElems;
  _Float16* Wqh = Xvh + xElems;
  _Float16* Wkh = Wqh + wElems;
  _Float16* Wvh = Wkh + wElems;
  _Float16* Woh = Wvh + wElems;
  _Float16* Qh  = Woh + wElems;
  _Float16* Kh  = Qh + xElems;
  _Float16* Vt  = Kh + xElems;
  _Float16* AO  = Vt + xElems;                        // [B,S,D] f16 context

  // 0) one-shot fp32->f16 conversion of activations and weights
  const int xBlocks = (int)(xElems / 8 / 256);        // 2048
  const int wBlocks = (int)(wElems / 8 / 256);        // 512
  cvt_f32_to_f16_kernel<<<xBlocks, 256, 0, stream>>>(query, Xqh, (int)xElems);
  cvt_f32_to_f16_kernel<<<xBlocks, 256, 0, stream>>>(key,   Xkh, (int)xElems);
  cvt_f32_to_f16_kernel<<<xBlocks, 256, 0, stream>>>(value, Xvh, (int)xElems);
  cvt_f32_to_f16_kernel<<<wBlocks, 256, 0, stream>>>(Wq, Wqh, (int)wElems);
  cvt_f32_to_f16_kernel<<<wBlocks, 256, 0, stream>>>(Wk, Wkh, (int)wElems);
  cvt_f32_to_f16_kernel<<<wBlocks, 256, 0, stream>>>(Wv, Wvh, (int)wElems);
  cvt_f32_to_f16_kernel<<<wBlocks, 256, 0, stream>>>(Wo, Woh, (int)wElems);

  // 1) QKV projections: 2048 wave-tiles (32x64) per matrix, 4 waves/block
  qkv_proj_kernel<<<dim3(512, 3, 1), 128, 0, stream>>>(Xqh, Xkh, Xvh,
                                                       Wqh, Wkh, Wvh, Qh, Kh, Vt);
  // 2) attention: one WG per (b,h,16-query tile) = 2*16*128 = 4096 WGs
  attention_kernel<<<dim3(4096), 128, SMEM_TOTAL, stream>>>(Qh, Kh, Vt, mask, attnw, AO);
  // 3) output projection
  out_proj_kernel<<<dim3(512), 128, 0, stream>>>(AO, Woh, bo, out);
}
